// LittleBitLinear_75127567941708
// MI455X (gfx1250) — compile-verified, matrix operand order checked by reading, not run on hardware
//
#include <hip/hip_runtime.h>
#include <hip/hip_bf16.h>

typedef __attribute__((ext_vector_type(16))) _Float16 v16h;
typedef __attribute__((ext_vector_type(8)))  _Float16 v8h;
typedef __attribute__((ext_vector_type(8)))  float    v8f;

#define DIM_IN   4096
#define DIM_OUT  4096
#define RANK     256
#define SCAT     512      // 2 * RANK (both branches concatenated)
#define NTOK     8192     // B * T
#define BM       128
#define BN       128
#define KB       64       // K-halves per LDS stage (2 WMMA k-steps)
#define LDT      72       // padded LDS row stride in halves (64 data + 8 pad)

__device__ __forceinline__ float sgnf(float v) {
    return (v > 0.f) ? 1.f : (v < 0.f ? -1.f : 0.f);
}

// WV[s_cat, i] = sign(V_b[s,i]) * v2_b[i] * v1_b[s] * u2_b[s]   (f16, row-major [512,4096])
__global__ __launch_bounds__(256) void prep_wv_kernel(
    const float* __restrict__ V,   const float* __restrict__ V_R,
    const float* __restrict__ v1,  const float* __restrict__ v2,
    const float* __restrict__ u2,  const float* __restrict__ v1_R,
    const float* __restrict__ v2_R,const float* __restrict__ u2_R,
    _Float16* __restrict__ WV)
{
    int idx = blockIdx.x * 256 + threadIdx.x;
    if (idx >= SCAT * DIM_IN) return;
    int sc = idx >> 12;          // s_cat 0..511
    int i  = idx & (DIM_IN - 1);
    int s  = sc & (RANK - 1);
    bool rb = sc >= RANK;
    const float* Vp = rb ? V_R : V;
    float sg = sgnf(Vp[(size_t)s * DIM_IN + i]);
    float sc1 = rb ? v1_R[s] : v1[s];
    float sc2 = rb ? v2_R[i] : v2[i];
    float sc3 = rb ? u2_R[s] : u2[s];
    WV[idx] = (_Float16)(sg * sc1 * sc2 * sc3);
}

// WU[o, s_cat] = sign(U_b[o,s]) * u1_b[o]   (f16, row-major [4096,512])
__global__ __launch_bounds__(256) void prep_wu_kernel(
    const float* __restrict__ U, const float* __restrict__ U_R,
    const float* __restrict__ u1, const float* __restrict__ u1_R,
    _Float16* __restrict__ WU)
{
    int idx = blockIdx.x * 256 + threadIdx.x;
    if (idx >= DIM_OUT * SCAT) return;
    int o  = idx >> 9;           // 0..4095
    int sc = idx & (SCAT - 1);
    int s  = sc & (RANK - 1);
    bool rb = sc >= RANK;
    float sg = sgnf((rb ? U_R : U)[(size_t)o * RANK + s]);
    WU[idx] = (_Float16)(sg * (rb ? u1_R[o] : u1[o]));
}

// ---- GEMM1: H[8192,512] (f16) = X[8192,4096] (f32) @ WV[512,4096]^T (f16) ----
// Register-staged loads (A needs f32->f16 conversion in-flight).
__global__ __launch_bounds__(256) void gemm1_kernel(
    const float* __restrict__ X, const _Float16* __restrict__ WV,
    _Float16* __restrict__ H)
{
    __shared__ _Float16 As[BM * LDT];
    __shared__ _Float16 Bs[BN * LDT];

    const int tid  = threadIdx.x;
    const int m0   = blockIdx.y * BM;
    const int n0   = blockIdx.x * BN;
    const int lane = tid & 31;
    const int wave = tid >> 5;
    const int wm   = wave >> 2;          // 0..1  (64 rows each)
    const int wn   = wave & 3;           // 0..3  (32 cols each)
    const int rr   = lane & 15;
    const int hs   = lane >> 4;

    const int lrow = tid >> 1;           // 0..127
    const int lcol = (tid & 1) * 32;     // 0 or 32

    const float*    xg = X  + (size_t)(m0 + lrow) * DIM_IN + lcol;
    const _Float16* bg = WV + (size_t)(n0 + lrow) * DIM_IN + lcol;

    float4 a_regs[8];
    v8h    b_regs[4];

    auto load_tile = [&](int kb) {
        const float* ap = xg + kb * KB;
#pragma unroll
        for (int j = 0; j < 8; ++j) a_regs[j] = *(const float4*)(ap + j * 4);
        const _Float16* bp = bg + kb * KB;
#pragma unroll
        for (int j = 0; j < 4; ++j) b_regs[j] = *(const v8h*)(bp + j * 8);
    };

    auto store_tile = [&]() {
#pragma unroll
        for (int j = 0; j < 4; ++j) {
            v8h h;
            h[0] = (_Float16)a_regs[2*j].x;   h[1] = (_Float16)a_regs[2*j].y;
            h[2] = (_Float16)a_regs[2*j].z;   h[3] = (_Float16)a_regs[2*j].w;
            h[4] = (_Float16)a_regs[2*j+1].x; h[5] = (_Float16)a_regs[2*j+1].y;
            h[6] = (_Float16)a_regs[2*j+1].z; h[7] = (_Float16)a_regs[2*j+1].w;
            *(v8h*)(&As[lrow * LDT + lcol + j * 8]) = h;
            *(v8h*)(&Bs[lrow * LDT + lcol + j * 8]) = b_regs[j];
        }
    };

    auto frag = [&](const _Float16* tbase, int kk) -> v16h {
        const _Float16* p = tbase + rr * LDT + kk * 32 + hs * 8;
        v16h f;
        ((v8h*)&f)[0] = *(const v8h*)p;
        ((v8h*)&f)[1] = *(const v8h*)(p + 16);
        return f;
    };

    v8f acc[4][2] = {};
    const int nk = DIM_IN / KB;   // 64
    load_tile(0);
    for (int kb = 0; kb < nk; ++kb) {
        __syncthreads();
        store_tile();
        __syncthreads();
        if (kb + 1 < nk) {
            load_tile(kb + 1);
            if (kb + 2 < nk) {
                __builtin_prefetch(xg + (size_t)(kb + 2) * KB, 0, 1);
                __builtin_prefetch(bg + (size_t)(kb + 2) * KB, 0, 1);
            }
        }
#pragma unroll
        for (int kk = 0; kk < 2; ++kk) {
            v16h bf0 = frag(&Bs[(wn * 32 +  0) * LDT], kk);
            v16h bf1 = frag(&Bs[(wn * 32 + 16) * LDT], kk);
            v16h af0 = frag(&As[(wm * 64 +  0) * LDT], kk);
            v16h af1 = frag(&As[(wm * 64 + 16) * LDT], kk);
            v16h af2 = frag(&As[(wm * 64 + 32) * LDT], kk);
            v16h af3 = frag(&As[(wm * 64 + 48) * LDT], kk);
            acc[0][0] = __builtin_amdgcn_wmma_f32_16x16x32_f16(false, af0, false, bf0, (short)0, acc[0][0], false, false);
            acc[0][1] = __builtin_amdgcn_wmma_f32_16x16x32_f16(false, af0, false, bf1, (short)0, acc[0][1], false, false);
            acc[1][0] = __builtin_amdgcn_wmma_f32_16x16x32_f16(false, af1, false, bf0, (short)0, acc[1][0], false, false);
            acc[1][1] = __builtin_amdgcn_wmma_f32_16x16x32_f16(false, af1, false, bf1, (short)0, acc[1][1], false, false);
            acc[2][0] = __builtin_amdgcn_wmma_f32_16x16x32_f16(false, af2, false, bf0, (short)0, acc[2][0], false, false);
            acc[2][1] = __builtin_amdgcn_wmma_f32_16x16x32_f16(false, af2, false, bf1, (short)0, acc[2][1], false, false);
            acc[3][0] = __builtin_amdgcn_wmma_f32_16x16x32_f16(false, af3, false, bf0, (short)0, acc[3][0], false, false);
            acc[3][1] = __builtin_amdgcn_wmma_f32_16x16x32_f16(false, af3, false, bf1, (short)0, acc[3][1], false, false);
        }
    }

#pragma unroll
    for (int mi = 0; mi < 4; ++mi) {
        int m = m0 + wm * 64 + mi * 16 + hs * 8;
#pragma unroll
        for (int ni = 0; ni < 2; ++ni) {
            int n = n0 + wn * 32 + ni * 16 + rr;
            _Float16* hp = H + (size_t)m * SCAT + n;
#pragma unroll
            for (int r = 0; r < 8; ++r)
                hp[(size_t)r * SCAT] = (_Float16)acc[mi][ni][r];
        }
    }
}

// ---- GEMM2: Y[8192,4096] (f32) = H[8192,512] @ WU[4096,512]^T + bias ----
// Async global->LDS DMA path (ASYNCcnt) with double-buffered LDS.
__global__ __launch_bounds__(256) void gemm2_kernel(
    const _Float16* __restrict__ H, const _Float16* __restrict__ WU,
    const float* __restrict__ bias, float* __restrict__ Y)
{
    __shared__ _Float16 As[2][BM * LDT];
    __shared__ _Float16 Bs[2][BN * LDT];

    const int tid  = threadIdx.x;
    const int m0   = blockIdx.y * BM;
    const int n0   = blockIdx.x * BN;
    const int lane = tid & 31;
    const int wave = tid >> 5;
    const int wm   = wave >> 2;
    const int wn   = wave & 3;
    const int rr   = lane & 15;
    const int hs   = lane >> 4;

    const int lrow = tid >> 1;
    const int lcol = (tid & 1) * 32;

    const _Float16* ag = H  + (size_t)(m0 + lrow) * SCAT + lcol;
    const _Float16* bg = WU + (size_t)(n0 + lrow) * SCAT + lcol;

    // hoisted 32-bit LDS destination addresses (hardware maps flat->LDS via addr[31:0])
    const unsigned laddrA[2] = {
        (unsigned)(size_t)(void*)&As[0][lrow * LDT + lcol],
        (unsigned)(size_t)(void*)&As[1][lrow * LDT + lcol] };
    const unsigned laddrB[2] = {
        (unsigned)(size_t)(void*)&Bs[0][lrow * LDT + lcol],
        (unsigned)(size_t)(void*)&Bs[1][lrow * LDT + lcol] };

    // Issue 8 async 16B DMA copies (4 A + 4 B) for tile kb into buffer `buf`.
    // ISA: INST_OFFSET is added to BOTH the LDS and global addresses, so one
    // base LDS VGPR + one base global pair + literal offsets covers the tile.
    auto async_tile = [&](int kb, int buf) {
        const _Float16* ap = ag + kb * KB;
        const _Float16* bp = bg + kb * KB;
        unsigned la = laddrA[buf];
        unsigned lb = laddrB[buf];
        asm volatile("global_load_async_to_lds_b128 %0, %1, off"
                     :: "v"(la), "v"(ap) : "memory");
        asm volatile("global_load_async_to_lds_b128 %0, %1, off offset:16"
                     :: "v"(la), "v"(ap) : "memory");
        asm volatile("global_load_async_to_lds_b128 %0, %1, off offset:32"
                     :: "v"(la), "v"(ap) : "memory");
        asm volatile("global_load_async_to_lds_b128 %0, %1, off offset:48"
                     :: "v"(la), "v"(ap) : "memory");
        asm volatile("global_load_async_to_lds_b128 %0, %1, off"
                     :: "v"(lb), "v"(bp) : "memory");
        asm volatile("global_load_async_to_lds_b128 %0, %1, off offset:16"
                     :: "v"(lb), "v"(bp) : "memory");
        asm volatile("global_load_async_to_lds_b128 %0, %1, off offset:32"
                     :: "v"(lb), "v"(bp) : "memory");
        asm volatile("global_load_async_to_lds_b128 %0, %1, off offset:48"
                     :: "v"(lb), "v"(bp) : "memory");
    };

    auto frag = [&](const _Float16* tbase, int kk) -> v16h {
        const _Float16* p = tbase + rr * LDT + kk * 32 + hs * 8;
        v16h f;
        ((v8h*)&f)[0] = *(const v8h*)p;
        ((v8h*)&f)[1] = *(const v8h*)(p + 16);
        return f;
    };

    v8f acc[4][2] = {};
    const int nk = SCAT / KB;     // 8

    async_tile(0, 0);
    asm volatile("s_wait_asynccnt 0x0" ::: "memory");
    __syncthreads();

#pragma unroll
    for (int kb = 0; kb < nk; ++kb) {
        // prefetch next tile into the other buffer (its last readers finished
        // before the barrier that opened this iteration)
        if (kb + 1 < nk) async_tile(kb + 1, (kb + 1) & 1);

        const _Float16* Ab = &As[kb & 1][0];
        const _Float16* Bb = &Bs[kb & 1][0];
#pragma unroll
        for (int kk = 0; kk < 2; ++kk) {
            v16h bf0 = frag(Bb + (wn * 32 +  0) * LDT, kk);
            v16h bf1 = frag(Bb + (wn * 32 + 16) * LDT, kk);
            v16h af0 = frag(Ab + (wm * 64 +  0) * LDT, kk);
            v16h af1 = frag(Ab + (wm * 64 + 16) * LDT, kk);
            v16h af2 = frag(Ab + (wm * 64 + 32) * LDT, kk);
            v16h af3 = frag(Ab + (wm * 64 + 48) * LDT, kk);
            acc[0][0] = __builtin_amdgcn_wmma_f32_16x16x32_f16(false, af0, false, bf0, (short)0, acc[0][0], false, false);
            acc[0][1] = __builtin_amdgcn_wmma_f32_16x16x32_f16(false, af0, false, bf1, (short)0, acc[0][1], false, false);
            acc[1][0] = __builtin_amdgcn_wmma_f32_16x16x32_f16(false, af1, false, bf0, (short)0, acc[1][0], false, false);
            acc[1][1] = __builtin_amdgcn_wmma_f32_16x16x32_f16(false, af1, false, bf1, (short)0, acc[1][1], false, false);
            acc[2][0] = __builtin_amdgcn_wmma_f32_16x16x32_f16(false, af2, false, bf0, (short)0, acc[2][0], false, false);
            acc[2][1] = __builtin_amdgcn_wmma_f32_16x16x32_f16(false, af2, false, bf1, (short)0, acc[2][1], false, false);
            acc[3][0] = __builtin_amdgcn_wmma_f32_16x16x32_f16(false, af3, false, bf0, (short)0, acc[3][0], false, false);
            acc[3][1] = __builtin_amdgcn_wmma_f32_16x16x32_f16(false, af3, false, bf1, (short)0, acc[3][1], false, false);
        }

        // own async ops complete in order: draining to 0 means next tile landed
        asm volatile("s_wait_asynccnt 0x0" ::: "memory");
        __syncthreads();
    }

#pragma unroll
    for (int mi = 0; mi < 4; ++mi) {
        int m = m0 + wm * 64 + mi * 16 + hs * 8;
#pragma unroll
        for (int ni = 0; ni < 2; ++ni) {
            int n = n0 + wn * 32 + ni * 16 + rr;
            float bv = bias[n];
            float* yp = Y + (size_t)m * DIM_OUT + n;
#pragma unroll
            for (int r = 0; r < 8; ++r)
                yp[(size_t)r * DIM_OUT] = acc[mi][ni][r] + bv;
        }
    }
}

extern "C" void kernel_launch(void* const* d_in, const int* in_sizes, int n_in,
                              void* d_out, int out_size, void* d_ws, size_t ws_size,
                              hipStream_t stream) {
    (void)in_sizes; (void)n_in; (void)out_size; (void)ws_size;
    const float* x    = (const float*)d_in[0];
    const float* V    = (const float*)d_in[1];
    const float* U    = (const float*)d_in[2];
    const float* v1   = (const float*)d_in[3];
    const float* v2   = (const float*)d_in[4];
    const float* u1   = (const float*)d_in[5];
    const float* u2   = (const float*)d_in[6];
    const float* V_R  = (const float*)d_in[7];
    const float* U_R  = (const float*)d_in[8];
    const float* v1_R = (const float*)d_in[9];
    const float* v2_R = (const float*)d_in[10];
    const float* u1_R = (const float*)d_in[11];
    const float* u2_R = (const float*)d_in[12];
    const float* bias = (const float*)d_in[13];

    _Float16* WV = (_Float16*)d_ws;                      // [512, 4096]  4 MB
    _Float16* WU = WV + (size_t)SCAT * DIM_IN;           // [4096, 512]  4 MB
    _Float16* H  = WU + (size_t)DIM_OUT * SCAT;          // [8192, 512]  8 MB
    float* Y = (float*)d_out;

    prep_wv_kernel<<<(SCAT * DIM_IN + 255) / 256, 256, 0, stream>>>(
        V, V_R, v1, v2, u2, v1_R, v2_R, u2_R, WV);
    prep_wu_kernel<<<(DIM_OUT * SCAT + 255) / 256, 256, 0, stream>>>(
        U, U_R, u1, u1_R, WU);
    gemm1_kernel<<<dim3(SCAT / BN, NTOK / BM), 256, 0, stream>>>(x, WV, H);
    gemm2_kernel<<<dim3(DIM_OUT / BN, NTOK / BM), 256, 0, stream>>>(H, WU, bias, Y);
}
